// TopKSparseAutoencoder_17394617549179
// MI455X (gfx1250) — matrix operand order repaired and checked by reading, not compiled, not run on hardware
//
#include <hip/hip_runtime.h>
#include <hip/hip_bf16.h>

#define INPUT_DIM  768
#define HIDDEN_DIM 24576
#define TOPK       50

// ---- CDNA5 WMMA vector types (wave32) ----
typedef __attribute__((ext_vector_type(16))) __bf16 bf16x16;
typedef __attribute__((ext_vector_type(8)))  __bf16 bf16x8;
typedef __attribute__((ext_vector_type(8)))  float  f32x8;

// ---- TDM descriptor group types (probe-confirmed builtin arg types) ----
typedef __attribute__((ext_vector_type(4))) unsigned tdm_g0_t;
typedef __attribute__((ext_vector_type(8))) int      tdm_g1_t;
typedef __attribute__((ext_vector_type(4))) int      tdm_g2_t;

#if __has_builtin(__builtin_amdgcn_tensor_load_to_lds) && \
    __has_builtin(__builtin_amdgcn_tensor_store_from_lds)
#define HAVE_TDM 1
#else
#define HAVE_TDM 0
#endif

__device__ __forceinline__ unsigned short f32_to_bf16_rne(float f) {
    unsigned u = __float_as_uint(f);
    u += 0x7FFFu + ((u >> 16) & 1u);   // round-to-nearest-even
    return (unsigned short)(u >> 16);
}

#if HAVE_TDM
// Build a D# for a 1-row 2D tensor of n f32 (tile == whole row) and issue
// TENSOR_LOAD_TO_LDS / TENSOR_STORE_FROM_LDS (CDNA5 ISA).
__device__ __forceinline__ void tdm_make_desc(tdm_g0_t& g0, tdm_g1_t& g1,
                                              const void* lds_ptr,
                                              const void* global_ptr, int n)
{
    unsigned lds_off = (unsigned)(unsigned long long)lds_ptr;     // low 32b = LDS offset
    unsigned long long ga = (unsigned long long)global_ptr;
    g0[0] = 1u;                                                   // count=1, user D#
    g0[1] = lds_off;                                              // lds_addr
    g0[2] = (unsigned)ga;                                         // global_addr[31:0]
    g0[3] = (unsigned)((ga >> 32) & 0x1FFFFFFu) | (2u << 30);     // addr[56:32] | type=2
    g1 = tdm_g1_t{};
    g1[0] = (int)(2u << 16);                                      // data_size=4B, mask=0
    g1[1] = (int)(((unsigned)n & 0xFFFFu) << 16);                 // tensor_dim0[15:0]
    g1[2] = (int)((((unsigned)n >> 16) & 0xFFFFu) | (1u << 16));  // dim0[31:16] | tensor_dim1=1
    g1[3] = (int)(((unsigned)n & 0xFFFFu) << 16);                 // tile_dim0 = n
    g1[4] = 1;                                                    // tile_dim1=1, tile_dim2=0
    g1[5] = n;                                                    // tensor_dim0_stride[31:0]
    g1[6] = 0; g1[7] = 0;
}

__device__ __forceinline__ void tdm_row_load(float* lds_dst, const float* gsrc, int n) {
    tdm_g0_t g0; tdm_g1_t g1;
    tdm_make_desc(g0, g1, lds_dst, gsrc, n);
    tdm_g2_t g2 = {}; tdm_g2_t g3 = {};
#if __clang_major__ >= 23
    tdm_g1_t g4 = {};
    __builtin_amdgcn_tensor_load_to_lds(g0, g1, g2, g3, g4, 0);
#else
    __builtin_amdgcn_tensor_load_to_lds(g0, g1, g2, g3, 0);
#endif
}

__device__ __forceinline__ void tdm_row_store(const float* lds_src, float* gdst, int n) {
    tdm_g0_t g0; tdm_g1_t g1;
    tdm_make_desc(g0, g1, lds_src, gdst, n);
    tdm_g2_t g2 = {}; tdm_g2_t g3 = {};
#if __clang_major__ >= 23
    tdm_g1_t g4 = {};
    __builtin_amdgcn_tensor_store_from_lds(g0, g1, g2, g3, g4, 0);
#else
    __builtin_amdgcn_tensor_store_from_lds(g0, g1, g2, g3, 0);
#endif
}
#endif  // HAVE_TDM

// =====================================================================
// Kernel 1: acts = relu((x - pre_bias) @ W_enc^T + latent_bias)
// Tiling: BM=64 x BN=64, K-step 32 (one bf16 WMMA K per LDS stage).
// 128 threads = 4 wave32; each wave owns a 32x32 block = 2x2 WMMA tiles.
// =====================================================================
#define BM 64
#define BN 64
#define BK 32
#define LDT 40   // padded LDS row stride in ushort (80 B, 16B-aligned frags)

__global__ __launch_bounds__(128)
void encoder_gemm_relu(const float* __restrict__ x,
                       const float* __restrict__ W_enc,
                       const float* __restrict__ pre_bias,
                       const float* __restrict__ latent_bias,
                       float* __restrict__ acts, int B)
{
    __shared__ unsigned short As[BM * LDT];   // x tile, bf16
    __shared__ unsigned short Bs[BN * LDT];   // W_enc tile, bf16

    const int tid  = threadIdx.x;
    const int lane = tid & 31;
    const int wave = tid >> 5;          // 0..3
    const int wm   = wave >> 1;         // wave row (0..1)
    const int wn   = wave & 1;          // wave col (0..1)
    const int half = lane >> 4;         // lane half (0/1)
    const int l16  = lane & 15;

    const int rowBase = blockIdx.y * BM;
    const int colBase = blockIdx.x * BN;

    f32x8 acc[2][2] = {};

    for (int k0 = 0; k0 < INPUT_DIM; k0 += BK) {
        // ---- stage A tile: (x - pre_bias) -> bf16 LDS ----
        #pragma unroll
        for (int i = 0; i < 4; ++i) {
            int q  = tid + i * 128;       // 0..511 float4 slots
            int r  = q >> 3;              // tile row 0..63
            int kq = (q & 7) << 2;        // 0,4,...,28
            float4 v = make_float4(0.f, 0.f, 0.f, 0.f);
            int gr = rowBase + r;
            if (gr < B)
                v = *(const float4*)(x + (size_t)gr * INPUT_DIM + k0 + kq);
            float4 pb = *(const float4*)(pre_bias + k0 + kq);
            unsigned short* dst = &As[r * LDT + kq];
            dst[0] = f32_to_bf16_rne(v.x - pb.x);
            dst[1] = f32_to_bf16_rne(v.y - pb.y);
            dst[2] = f32_to_bf16_rne(v.z - pb.z);
            dst[3] = f32_to_bf16_rne(v.w - pb.w);
        }
        // ---- stage B tile: W_enc[n, k] -> bf16 LDS ----
        #pragma unroll
        for (int i = 0; i < 4; ++i) {
            int q  = tid + i * 128;
            int r  = q >> 3;
            int kq = (q & 7) << 2;
            float4 v = *(const float4*)(W_enc + (size_t)(colBase + r) * INPUT_DIM + k0 + kq);
            unsigned short* dst = &Bs[r * LDT + kq];
            dst[0] = f32_to_bf16_rne(v.x);
            dst[1] = f32_to_bf16_rne(v.y);
            dst[2] = f32_to_bf16_rne(v.z);
            dst[3] = f32_to_bf16_rne(v.w);
        }
        // prefetch next K-slab (emits global_prefetch_b8)
        if (k0 + BK < INPUT_DIM) {
            __builtin_prefetch(x     + (size_t)(rowBase + (tid >> 1)) * INPUT_DIM + k0 + BK, 0, 1);
            __builtin_prefetch(W_enc + (size_t)(colBase + (tid >> 1)) * INPUT_DIM + k0 + BK, 0, 1);
        }
        __syncthreads();

        // ---- fragments per documented CDNA5 16-bit layouts + WMMA ----
        #pragma unroll
        for (int tm = 0; tm < 2; ++tm) {
            int ar = wm * 32 + tm * 16 + l16;   // A row M = l16
            // lane-half 0: K=0..7 & 16..23 ; half 1: K=8..15 & 24..31
            bf16x8 alo = *(const bf16x8*)&As[ar * LDT + half * 8];
            bf16x8 ahi = *(const bf16x8*)&As[ar * LDT + half * 8 + 16];
            bf16x16 a = __builtin_shufflevector(alo, ahi,
                          0,1,2,3,4,5,6,7,8,9,10,11,12,13,14,15);
            #pragma unroll
            for (int tn = 0; tn < 2; ++tn) {
                int bn = wn * 32 + tn * 16 + l16;   // B col N = l16
                // lane-half 0: K=0..7 & 16..23 ; half 1: K=8..15 & 24..31
                bf16x8 blo = *(const bf16x8*)&Bs[bn * LDT + half * 8];
                bf16x8 bhi = *(const bf16x8*)&Bs[bn * LDT + half * 8 + 16];
                bf16x16 bb = __builtin_shufflevector(blo, bhi,
                               0,1,2,3,4,5,6,7,8,9,10,11,12,13,14,15);
                acc[tm][tn] = __builtin_amdgcn_wmma_f32_16x16x32_bf16(
                    false, a, false, bb, (short)0, acc[tm][tn], false, false);
            }
        }
        __syncthreads();
    }

    // ---- epilogue: + latent_bias, relu, store dense acts ----
    #pragma unroll
    for (int tm = 0; tm < 2; ++tm) {
        #pragma unroll
        for (int tn = 0; tn < 2; ++tn) {
            int gc = colBase + wn * 32 + tn * 16 + l16;
            float lb = latent_bias[gc];
            #pragma unroll
            for (int r = 0; r < 8; ++r) {
                int gr = rowBase + wm * 32 + tm * 16 + r + half * 8;  // C layout
                if (gr < B) {
                    float v = acc[tm][tn][r] + lb;
                    acts[(size_t)gr * HIDDEN_DIM + gc] = v > 0.f ? v : 0.f;
                }
            }
        }
    }
}

// =====================================================================
// Kernel 2: per-row top-K via bitwise binary search on the row staged in
// LDS (96 KB — fits CDNA5's 320 KB WGP LDS). Row staging in/out is done
// with the Tensor Data Mover (TENSOR_LOAD_TO_LDS / TENSOR_STORE_FROM_LDS,
// tracked on TENSORcnt). Post-ReLU values are >= 0, so float ordering ==
// uint bit ordering.
// =====================================================================
__global__ __launch_bounds__(256)
void topk_sparsify(float* __restrict__ acts,
                   float* __restrict__ topkV,
                   float* __restrict__ topkI)
{
    extern __shared__ float row[];              // HIDDEN_DIM floats
    __shared__ unsigned s_cnt;
    __shared__ unsigned s_pos;
    __shared__ float    candV[64];
    __shared__ int      candI[64];
    __shared__ int      tieIdx[TOPK + 2];
    __shared__ int      s_tieCnt;

    const int tid = threadIdx.x;
    const int nt  = blockDim.x;
    const size_t b = blockIdx.x;
    float* g = acts + b * (size_t)HIDDEN_DIM;

#if HAVE_TDM
    // one TDM descriptor pulls the whole 96KB row into LDS (ASYNC, TENSORcnt)
    if (tid == 0) tdm_row_load(row, g, HIDDEN_DIM);
    if (tid == 0) s_cnt = 0;
    if (tid < 32) __builtin_amdgcn_s_wait_tensorcnt(0);
    __syncthreads();
#else
    for (int i = tid; i < HIDDEN_DIM; i += nt) row[i] = g[i];
    if (tid == 0) s_cnt = 0;
    __syncthreads();
#endif

    // binary search for the K-th largest value's bit pattern
    unsigned lo = 0u, hi = 0x7F800000u;
    while (lo < hi) {
        unsigned mid = lo + ((hi - lo + 1u) >> 1);
        unsigned c = 0;
        for (int i = tid; i < HIDDEN_DIM; i += nt)
            c += (__float_as_uint(row[i]) >= mid) ? 1u : 0u;
        #pragma unroll
        for (int off = 16; off > 0; off >>= 1)
            c += __shfl_down(c, off, 32);
        if ((tid & 31) == 0 && c) atomicAdd(&s_cnt, c);
        __syncthreads();
        unsigned total = s_cnt;
        __syncthreads();
        if (tid == 0) s_cnt = 0;
        __syncthreads();
        if (total >= TOPK) lo = mid; else hi = mid - 1u;
    }
    const unsigned thr  = lo;
    const float    thrF = __uint_as_float(thr);

    // collect strictly-greater candidates (count < K by construction)
    if (tid == 0) s_pos = 0;
    __syncthreads();
    for (int i = tid; i < HIDDEN_DIM; i += nt) {
        unsigned u = __float_as_uint(row[i]);
        if (u > thr) {
            unsigned p = atomicAdd(&s_pos, 1u);
            if (p < 64u) { candV[p] = row[i]; candI[p] = i; }
        }
    }
    __syncthreads();

    if (tid == 0) {
        int ng = (int)s_pos; if (ng > TOPK) ng = TOPK;
        // fill ties by smallest index (deterministic, matches lax.top_k)
        int need = TOPK - ng, tc = 0;
        for (int i = 0; i < HIDDEN_DIM && need > 0; ++i) {
            if (__float_as_uint(row[i]) == thr) {
                candV[ng + tc] = thrF;
                candI[ng + tc] = i;
                tieIdx[tc] = i;
                ++tc; --need;
            }
        }
        s_tieCnt = tc;
        // selection sort: value desc, index asc (K=50 -> 2.5k cmps)
        for (int i = 0; i < TOPK - 1; ++i) {
            int best = i;
            for (int j = i + 1; j < TOPK; ++j)
                if (candV[j] > candV[best] ||
                    (candV[j] == candV[best] && candI[j] < candI[best])) best = j;
            if (best != i) {
                float fv = candV[i]; candV[i] = candV[best]; candV[best] = fv;
                int   iv = candI[i]; candI[i] = candI[best]; candI[best] = iv;
            }
        }
    }
    __syncthreads();

    if (tid < TOPK) {
        topkV[b * TOPK + tid] = candV[tid];
        topkI[b * TOPK + tid] = (float)candI[tid];
    }

    // sparsify: mask the row in LDS, then one TDM store streams it out
    const int tieCnt = s_tieCnt;
    for (int i = tid; i < HIDDEN_DIM; i += nt) {
        float f = row[i];
        unsigned u = __float_as_uint(f);
        bool keep = (u > thr);
        if (!keep && u == thr)
            for (int t = 0; t < tieCnt; ++t)
                if (tieIdx[t] == i) { keep = true; break; }
#if HAVE_TDM
        row[i] = keep ? f : 0.0f;
#else
        g[i] = keep ? f : 0.0f;
#endif
    }
#if HAVE_TDM
    __syncthreads();                       // all LDS writes visible
    if (tid == 0) tdm_row_store(row, g, HIDDEN_DIM);
    if (tid < 32) __builtin_amdgcn_s_wait_tensorcnt(0);   // LDS must outlive DMA
#endif
}

// =====================================================================
// Kernel 3a: one-time W_dec transpose into workspace (coalesced decode).
// =====================================================================
__global__ __launch_bounds__(256)
void transpose_wdec(const float* __restrict__ Wd, float* __restrict__ WdT)
{
    __shared__ float t[32][33];
    const int hBase = blockIdx.x * 32;
    const int dBase = blockIdx.y * 32;
    const int tx = threadIdx.x & 31;
    const int ty = threadIdx.x >> 5;    // 0..7
    #pragma unroll
    for (int j = 0; j < 4; ++j) {
        int d = dBase + ty + j * 8;
        t[ty + j * 8][tx] = Wd[(size_t)d * HIDDEN_DIM + hBase + tx];
    }
    __syncthreads();
    #pragma unroll
    for (int j = 0; j < 4; ++j) {
        int h = hBase + ty + j * 8;
        WdT[(size_t)h * INPUT_DIM + dBase + tx] = t[tx][ty + j * 8];
    }
}

// =====================================================================
// Kernel 3b: reconstruction[b,:] = pre_bias + sum_k v_k * W_dec[:, idx_k]
// 50 nnz/row -> gather-FMA beats dense GEMM by ~500x in FLOPs.
// =====================================================================
__global__ __launch_bounds__(256)
void decode_sparse(const float* __restrict__ topkV,
                   const float* __restrict__ topkI,
                   const float* __restrict__ W,
                   const float* __restrict__ pre_bias,
                   float* __restrict__ rec, int transposed)
{
    __shared__ float sv[TOPK];
    __shared__ int   si[TOPK];
    const size_t b = blockIdx.x;
    const int tid = threadIdx.x;
    if (tid < TOPK) {
        sv[tid] = topkV[b * TOPK + tid];
        si[tid] = (int)topkI[b * TOPK + tid];
    }
    __syncthreads();

    float acc[3];
    #pragma unroll
    for (int j = 0; j < 3; ++j) acc[j] = pre_bias[tid + j * 256];

    for (int k = 0; k < TOPK; ++k) {
        float v = sv[k];
        int   h = si[k];
        if (transposed) {
            const float* wrow = W + (size_t)h * INPUT_DIM;
            #pragma unroll
            for (int j = 0; j < 3; ++j) acc[j] = fmaf(v, wrow[tid + j * 256], acc[j]);
        } else {
            #pragma unroll
            for (int j = 0; j < 3; ++j)
                acc[j] = fmaf(v, W[(size_t)(tid + j * 256) * HIDDEN_DIM + h], acc[j]);
        }
    }
    #pragma unroll
    for (int j = 0; j < 3; ++j) rec[b * INPUT_DIM + tid + j * 256] = acc[j];
}

// =====================================================================
extern "C" void kernel_launch(void* const* d_in, const int* in_sizes, int n_in,
                              void* d_out, int out_size, void* d_ws, size_t ws_size,
                              hipStream_t stream)
{
    const float* x           = (const float*)d_in[0];
    const float* W_enc       = (const float*)d_in[1];
    const float* W_dec       = (const float*)d_in[2];
    const float* pre_bias    = (const float*)d_in[3];
    const float* latent_bias = (const float*)d_in[4];
    const int B = in_sizes[0] / INPUT_DIM;   // 8192

    float* out    = (float*)d_out;
    float* rec    = out;                                    // [B, 768]
    float* sparse = rec    + (size_t)B * INPUT_DIM;         // [B, 24576]
    float* tv     = sparse + (size_t)B * HIDDEN_DIM;        // [B, 50]
    float* ti     = tv     + (size_t)B * TOPK;              // [B, 50]

    // 1) dense encoder GEMM (bf16 WMMA) -> relu acts written into sparse slot
    dim3 g1(HIDDEN_DIM / BN, (B + BM - 1) / BM);
    encoder_gemm_relu<<<g1, 128, 0, stream>>>(x, W_enc, pre_bias, latent_bias, sparse, B);

    // 2) per-row top-K + in-place sparsify (96 KB dynamic LDS, TDM staged)
    topk_sparsify<<<B, 256, HIDDEN_DIM * sizeof(float), stream>>>(sparse, tv, ti);

    // 3) sparse decode; coalesce via one-time W_dec transpose when ws allows
    const size_t wdT_bytes = (size_t)HIDDEN_DIM * INPUT_DIM * sizeof(float);
    int transposed = 0;
    const float* Wuse = W_dec;
    if (ws_size >= wdT_bytes) {
        float* WdT = (float*)d_ws;
        transpose_wdec<<<dim3(HIDDEN_DIM / 32, INPUT_DIM / 32), 256, 0, stream>>>(W_dec, WdT);
        Wuse = WdT;
        transposed = 1;
    }
    decode_sparse<<<B, 256, 0, stream>>>(tv, ti, Wuse, pre_bias, rec, transposed);
}
